// GCN_74397423501442
// MI455X (gfx1250) — matrix operand order repaired and checked by reading, not compiled, not run on hardware
//
#include <hip/hip_runtime.h>
#include <hip/hip_bf16.h>

#define NN 100000
#define NE 1600000

typedef __attribute__((ext_vector_type(16))) __bf16 v16bf;
typedef __attribute__((ext_vector_type(8)))  float  v8f;

__device__ __forceinline__ __bf16 f2bf(float f) { return (__bf16)f; }

// -------- weight packing into per-lane WMMA B-fragment layout ------------------
// Output: for fragment f = kt*NT + t, lane l, element j (16 bf16 per lane):
//   out[(f*32 + l)*16 + j] = W[kt*32 + (l>>4)*16 + j][t*16 + (l&15)]  (0 padded)
// in contiguous k order (k = 16*hi + j, same order as the A fragment), so a lane's whole fragment is one
// contiguous 32-byte chunk (two ds_load_b128 from LDS).
__global__ void pack_frag(const float* __restrict__ W, int K, int M,
                          __bf16* __restrict__ out, int KT, int NT) {
  int i = blockIdx.x * blockDim.x + threadIdx.x;
  int total = KT * NT * 512;
  if (i >= total) return;
  int j    = i & 15;
  int lane = (i >> 4) & 31;
  int f    = i >> 9;
  int tt   = f % NT, kt = f / NT;
  int hi = lane >> 4, l16 = lane & 15;
  int k = kt * 32 + hi * 16 + j;
  int m = tt * 16 + l16;
  float v = (k < K && m < M) ? W[k * M + m] : 0.0f;
  out[i] = f2bf(v);
}

// ---------------- degree / norm kernels ---------------------------------------
__global__ void deg_kernel(const long long* __restrict__ ei, const float* __restrict__ w,
                           float* __restrict__ deg, float* __restrict__ deg2, int E) {
  int e = blockIdx.x * blockDim.x + threadIdx.x;
  if (e >= E) return;
  int r = (int)ei[e], c = (int)ei[E + e];
  atomicAdd(deg + r, w[e]);
  atomicAdd(deg2 + c, 1.0f);
}

__global__ void dinv_kernel(const float* __restrict__ deg, const float* __restrict__ deg2,
                            float* __restrict__ dinv, float* __restrict__ dinv2, int n) {
  int i = blockIdx.x * blockDim.x + threadIdx.x;
  if (i >= n) return;
  float d = deg[i];
  dinv[i]  = d  > 0.f ? rsqrtf(fmaxf(d,  1e-30f)) : 0.f;
  float d2 = deg2[i];
  dinv2[i] = d2 > 0.f ? rsqrtf(fmaxf(d2, 1e-30f)) : 0.f;
}

__global__ void norm_kernel(const long long* __restrict__ ei, const float* __restrict__ w,
                            const float* __restrict__ dinv, const float* __restrict__ dinv2,
                            float* __restrict__ chebn, float* __restrict__ gcnn, int E) {
  int e = blockIdx.x * blockDim.x + threadIdx.x;
  if (e >= E) return;
  int r = (int)ei[e], c = (int)ei[E + e];
  chebn[e] = -dinv[r] * w[e] * dinv[c];
  gcnn[e]  = dinv2[r] * dinv2[c];
}

// ---------------- sparse propagation: out[col] += norm * z[row] ---------------
// F = 32 floats/node, 8 lanes per edge, float4 gathers, f32 atomic scatters.
__global__ void prop32(const long long* __restrict__ ei, const float* __restrict__ nrm,
                       const float* __restrict__ z, float* __restrict__ out, int E) {
  int tid = blockIdx.x * blockDim.x + threadIdx.x;
  int e = tid >> 3, sub = tid & 7;
  if (e >= E) return;
  if (sub == 0) {
    __builtin_prefetch(ei + e + 4096, 0, 0);        // edge stream ahead -> GL2
    __builtin_prefetch(ei + E + e + 4096, 0, 0);
    __builtin_prefetch(nrm + e + 4096, 0, 0);
  }
  int r = (int)ei[e], c = (int)ei[E + e];
  float w = nrm[e];
  float4 v = *((const float4*)(z + (size_t)r * 32) + sub);
  float* o = out + (size_t)c * 32 + sub * 4;
  atomicAdd(o + 0, w * v.x);
  atomicAdd(o + 1, w * v.y);
  atomicAdd(o + 2, w * v.z);
  atomicAdd(o + 3, w * v.w);
}

// F = 64 (padded), 16 lanes per edge; blockIdx.y = ARMA stack.
__global__ void prop64(const long long* __restrict__ ei, const float* __restrict__ nrm,
                       const float* __restrict__ z, float* __restrict__ out,
                       int E, size_t stackStride) {
  const float* zz = z + blockIdx.y * stackStride;
  float* oo = out + blockIdx.y * stackStride;
  int tid = blockIdx.x * blockDim.x + threadIdx.x;
  int e = tid >> 4, sub = tid & 15;
  if (e >= E) return;
  if (sub == 0) {
    __builtin_prefetch(ei + e + 4096, 0, 0);
    __builtin_prefetch(ei + E + e + 4096, 0, 0);
    __builtin_prefetch(nrm + e + 4096, 0, 0);
  }
  int r = (int)ei[e], c = (int)ei[E + e];
  float w = nrm[e];
  float4 v = *((const float4*)(zz + (size_t)r * 64) + sub);
  float* o = oo + (size_t)c * 64 + sub * 4;
  atomicAdd(o + 0, w * v.x);
  atomicAdd(o + 1, w * v.y);
  atomicAdd(o + 2, w * v.z);
  atomicAdd(o + 3, w * v.w);
}

// Tx2 = 2*agg - Tx0 (in place on agg)
__global__ void cheb_combine(float* __restrict__ agg, const float* __restrict__ tx0, int n) {
  int i = blockIdx.x * blockDim.x + threadIdx.x;
  if (i < n) agg[i] = 2.0f * agg[i] - tx0[i];
}

// feats[:,0:50] = relu(chebacc[:,0:50] + cheb_b)
__global__ void cheb_finish(const float* __restrict__ acc, const float* __restrict__ bias,
                            float* __restrict__ feats) {
  int i = blockIdx.x * blockDim.x + threadIdx.x;
  if (i >= NN * 64) return;
  int j = i & 63, n = i >> 6;
  if (j < 50) feats[(size_t)n * 128 + j] = fmaxf(acc[i] + bias[j], 0.0f);
}

// feats[:,50:100] = relu(mean over 2 stacks of armaOut)
__global__ void arma_mean(const float* __restrict__ a, float* __restrict__ feats) {
  int i = blockIdx.x * blockDim.x + threadIdx.x;
  if (i >= NN * 64) return;
  int j = i & 63, n = i >> 6;
  if (j < 50) {
    float v = 0.5f * (a[i] + a[(size_t)NN * 64 + i]);
    feats[(size_t)n * 128 + 50 + j] = fmaxf(v, 0.0f);
  }
}

// ---------------- generic WMMA GEMM: C = op(A@W [+C] [+add] [+bias]) ----------
// A: f32 [nrows, lda], converted to bf16 on the fly (16x32 A lane layout).
// Wp: bf16 fragments pre-packed per-lane (see pack_frag); staged once in LDS,
//     then each lane loads its 32-byte B fragment with two ds_load_b128.
// Each wave computes a 16-row slab x NT 16-col tiles via v_wmma_f32_16x16x32_bf16.
template <int KT, int NT>
__global__ void gemm_wmma(const float* __restrict__ A, int lda,
                          const __bf16* __restrict__ Wp,
                          float* __restrict__ C, int ldc, int ncols,
                          const float* __restrict__ addbuf, int ldadd,
                          const float* __restrict__ bias, int nbias,
                          int nrows, int beta, int do_relu) {
  constexpr int NELEM = KT * NT * 512;  // bf16 elements (= Kpad*Mpad)
  extern __shared__ __align__(16) __bf16 sW[];
  {
    const uint4* src = (const uint4*)Wp;
    uint4* dst = (uint4*)sW;
    for (int i = threadIdx.x; i < NELEM / 8; i += blockDim.x) dst[i] = src[i];
  }
  __syncthreads();

  const int wave = threadIdx.x >> 5;
  const int lane = threadIdx.x & 31;
  const int hi   = lane >> 4;
  const int l16  = lane & 15;
  const int row_base = (blockIdx.x * 8 + wave) * 16;
  if (row_base >= nrows) return;

  v8f acc[NT];
  const v8f vzero = {0.f, 0.f, 0.f, 0.f, 0.f, 0.f, 0.f, 0.f};
#pragma unroll
  for (int t = 0; t < NT; ++t) acc[t] = vzero;

  const v16bf* sFrag = (const v16bf*)sW;  // 32 bytes per lane-fragment
  const float* arow = A + (size_t)(row_base + l16) * lda;
#pragma unroll
  for (int kt = 0; kt < KT; ++kt) {
    const int k0 = kt * 32;
    // A fragment: lane (hi,l16) = row l16, K in [k0+hi*16 .. +15]
    v16bf afrag;
    const float* ap = arow + k0 + hi * 16;
#pragma unroll
    for (int j = 0; j < 8; ++j) {
      afrag[j]     = f2bf(ap[j]);
      afrag[8 + j] = f2bf(ap[8 + j]);
    }
#pragma unroll
    for (int t = 0; t < NT; ++t) {
      v16bf bfrag = sFrag[(kt * NT + t) * 32 + lane];
      acc[t] = __builtin_amdgcn_wmma_f32_16x16x32_bf16(
          false, afrag, false, bfrag, (short)0, acc[t], false, false);
    }
  }

  // C/D layout: element r, lanes 0-15 -> M=r, lanes 16-31 -> M=8+r; N=l16
#pragma unroll
  for (int t = 0; t < NT; ++t) {
#pragma unroll
    for (int r = 0; r < 8; ++r) {
      const int rr = row_base + hi * 8 + r;
      const int cc = t * 16 + l16;
      if (cc < ncols) {
        float v = acc[t][r];
        const size_t o = (size_t)rr * ldc + cc;
        if (beta) v += C[o];
        if (addbuf) v += addbuf[(size_t)rr * ldadd + cc];
        if (bias && cc < nbias) v += bias[cc];
        if (do_relu) v = fmaxf(v, 0.0f);
        C[o] = v;
      }
    }
  }
}

// ------------------------------------------------------------------------------
extern "C" void kernel_launch(void* const* d_in, const int* in_sizes, int n_in,
                              void* d_out, int out_size, void* d_ws, size_t ws_size,
                              hipStream_t stream) {
  (void)in_sizes; (void)n_in; (void)out_size; (void)ws_size;
  const float*     x     = (const float*)d_in[0];
  const long long* ei    = (const long long*)d_in[1];
  const float*     ew    = (const float*)d_in[2];
  const float*     W1    = (const float*)d_in[3];
  const float*     b1    = (const float*)d_in[4];
  const float*     chebw = (const float*)d_in[5];
  const float*     chebb = (const float*)d_in[6];
  const float*     initw = (const float*)d_in[7];
  const float*     armaw = (const float*)d_in[8];
  const float*     rootw = (const float*)d_in[9];
  const float*     armab = (const float*)d_in[10];
  const float*     W3    = (const float*)d_in[11];
  const float*     b3    = (const float*)d_in[12];
  float* out = (float*)d_out;

  const size_t N = NN, E = NE;
  char* ws = (char*)d_ws;
  size_t off = 0;
  auto alloc = [&](size_t bytes) -> char* {
    char* p = ws + off;
    off += (bytes + 255) & ~(size_t)255;
    return p;
  };
  float* deg     = (float*)alloc(N * 4);
  float* deg2    = (float*)alloc(N * 4);
  float* dinv    = (float*)alloc(N * 4);
  float* dinv2   = (float*)alloc(N * 4);
  float* chebn   = (float*)alloc(E * 4);
  float* gcnn    = (float*)alloc(E * 4);
  float* h       = (float*)alloc(N * 32 * 4);
  float* B0      = (float*)alloc(N * 32 * 4);
  float* B1      = (float*)alloc(N * 32 * 4);
  float* B2      = (float*)alloc(N * 32 * 4);
  float* chebacc = (float*)alloc(N * 64 * 4);
  float* armaT   = (float*)alloc(2 * N * 64 * 4);
  float* armaAgg = (float*)alloc(2 * N * 64 * 4);
  float* armaOut = (float*)alloc(2 * N * 64 * 4);
  float* feats   = (float*)alloc(N * 128 * 4);
  __bf16* wb_first = (__bf16*)alloc(128 * 32 * 2);
  __bf16* wb_cheb  = (__bf16*)alloc(7 * 32 * 64 * 2);
  __bf16* wb_init  = (__bf16*)alloc(2 * 32 * 64 * 2);
  __bf16* wb_arma  = (__bf16*)alloc(2 * 64 * 64 * 2);
  __bf16* wb_root  = (__bf16*)alloc(4 * 32 * 64 * 2);
  __bf16* wb_W3    = (__bf16*)alloc(128 * 32 * 2);

  auto packs = [&](const float* src, int K, int M, __bf16* dst, int KT, int NT) {
    int n = KT * NT * 512;
    pack_frag<<<(n + 255) / 256, 256, 0, stream>>>(src, K, M, dst, KT, NT);
  };
  packs(W1, 128, 32, wb_first, 4, 2);
  for (int k = 0; k < 7; ++k) packs(chebw + k * 32 * 50, 32, 50, wb_cheb + k * 32 * 64, 1, 4);
  for (int s = 0; s < 2; ++s) packs(initw + s * 32 * 50, 32, 50, wb_init + s * 32 * 64, 1, 4);
  for (int s = 0; s < 2; ++s) packs(armaw + s * 50 * 50, 50, 50, wb_arma + s * 64 * 64, 2, 4);
  for (int i = 0; i < 4; ++i) packs(rootw + i * 32 * 50, 32, 50, wb_root + i * 32 * 64, 1, 4);
  packs(W3, 100, 18, wb_W3, 4, 2);

  hipMemsetAsync(deg, 0, N * 4, stream);
  hipMemsetAsync(deg2, 0, N * 4, stream);
  hipMemsetAsync(feats, 0, N * 128 * 4, stream);

  deg_kernel<<<(int)((E + 255) / 256), 256, 0, stream>>>(ei, ew, deg, deg2, (int)E);
  dinv_kernel<<<(int)((N + 255) / 256), 256, 0, stream>>>(deg, deg2, dinv, dinv2, (int)N);
  norm_kernel<<<(int)((E + 255) / 256), 256, 0, stream>>>(ei, ew, dinv, dinv2, chebn, gcnn, (int)E);

  const int GBLK = ((int)(N / 16) + 7) / 8;  // 8 waves/block, 16 rows/wave
  const int PG32 = (int)((E * 8 + 255) / 256);
  const dim3 PG64((unsigned)((E * 16 + 255) / 256), 2);

  // h = relu(x @ W_first + b_first)
  gemm_wmma<4, 2><<<GBLK, 256, 128 * 32 * 2, stream>>>(
      x, 128, wb_first, h, 32, 32, nullptr, 0, b1, 32, (int)N, 0, 1);

  // ---- ChebConv ----
  gemm_wmma<1, 4><<<GBLK, 256, 32 * 64 * 2, stream>>>(
      h, 32, wb_cheb, chebacc, 64, 50, nullptr, 0, nullptr, 0, (int)N, 0, 0);
  hipMemsetAsync(B0, 0, N * 32 * 4, stream);
  prop32<<<PG32, 256, 0, stream>>>(ei, chebn, h, B0, (int)E);  // Tx1
  gemm_wmma<1, 4><<<GBLK, 256, 32 * 64 * 2, stream>>>(
      B0, 32, wb_cheb + 1 * 32 * 64, chebacc, 64, 50, nullptr, 0, nullptr, 0, (int)N, 1, 0);

  const float* Tx0 = h;
  float* Tx1 = B0;
  float* bufs[3] = {B1, B2, B0};
  for (int k = 2; k < 7; ++k) {
    float* agg = bufs[(k - 2) % 3];
    hipMemsetAsync(agg, 0, N * 32 * 4, stream);
    prop32<<<PG32, 256, 0, stream>>>(ei, chebn, Tx1, agg, (int)E);
    cheb_combine<<<(int)((N * 32 + 255) / 256), 256, 0, stream>>>(agg, Tx0, (int)(N * 32));
    gemm_wmma<1, 4><<<GBLK, 256, 32 * 64 * 2, stream>>>(
        agg, 32, wb_cheb + k * 32 * 64, chebacc, 64, 50, nullptr, 0, nullptr, 0, (int)N, 1, 0);
    Tx0 = Tx1;
    Tx1 = agg;
  }
  cheb_finish<<<(int)((N * 64 + 255) / 256), 256, 0, stream>>>(chebacc, chebb, feats);

  // ---- ARMAConv (2 stacks, 2 layers) ----
  const size_t SS = N * 64;  // stack stride (floats)
  for (int s = 0; s < 2; ++s)
    gemm_wmma<1, 4><<<GBLK, 256, 32 * 64 * 2, stream>>>(
        h, 32, wb_init + s * 32 * 64, armaT + s * SS, 64, 64,
        nullptr, 0, nullptr, 0, (int)N, 0, 0);
  hipMemsetAsync(armaAgg, 0, 2 * SS * 4, stream);
  prop64<<<PG64, 256, 0, stream>>>(ei, gcnn, armaT, armaAgg, (int)E, SS);
  for (int s = 0; s < 2; ++s)
    gemm_wmma<1, 4><<<GBLK, 256, 32 * 64 * 2, stream>>>(
        h, 32, wb_root + (0 * 2 + s) * 32 * 64, armaOut + s * SS, 64, 64,
        armaAgg + s * SS, 64, armab + (0 * 2 + s) * 50, 50, (int)N, 0, 1);

  for (int s = 0; s < 2; ++s)
    gemm_wmma<2, 4><<<GBLK, 256, 64 * 64 * 2, stream>>>(
        armaOut + s * SS, 64, wb_arma + s * 64 * 64, armaT + s * SS, 64, 64,
        nullptr, 0, nullptr, 0, (int)N, 0, 0);
  hipMemsetAsync(armaAgg, 0, 2 * SS * 4, stream);
  prop64<<<PG64, 256, 0, stream>>>(ei, gcnn, armaT, armaAgg, (int)E, SS);
  for (int s = 0; s < 2; ++s)
    gemm_wmma<1, 4><<<GBLK, 256, 32 * 64 * 2, stream>>>(
        h, 32, wb_root + (1 * 2 + s) * 32 * 64, armaOut + s * SS, 64, 64,
        armaAgg + s * SS, 64, armab + (1 * 2 + s) * 50, 50, (int)N, 0, 1);
  arma_mean<<<(int)((N * 64 + 255) / 256), 256, 0, stream>>>(armaOut, feats);

  // ---- classifier: out = feats @ W3 + b3 ----
  gemm_wmma<4, 2><<<GBLK, 256, 128 * 32 * 2, stream>>>(
      feats, 128, wb_W3, out, 18, 18, nullptr, 0, b3, 18, (int)N, 0, 0);
}
